// AtomicLinearTest_32555852104252
// MI455X (gfx1250) — matrix operand order repaired and checked by reading, not compiled, not run on hardware
//
#include <hip/hip_runtime.h>

#define TILE_M 128
#define TILE_N 128
#define TILE_K 32
#define MSIZE 4096
#define NSIZE 4096
#define KSIZE 4096
#define LDSS 40                    // 32 + 8 pad bf16 elems -> 80B row stride
#define TILE_BYTES (128 * LDSS * 2)  // 10240 B per padded tile

typedef __attribute__((ext_vector_type(16))) __bf16 v16bf;
typedef __attribute__((ext_vector_type(8)))  __bf16 v8bf;
typedef __attribute__((ext_vector_type(8)))  float  v8f;
typedef __attribute__((ext_vector_type(4)))  unsigned int u32x4;
typedef __attribute__((ext_vector_type(8)))  int i32x8;
typedef __attribute__((ext_vector_type(4)))  int i32x4;

static __device__ __forceinline__ unsigned short bf16_rn(float f) {
  unsigned int u = __float_as_uint(f);
  u += 0x7FFFu + ((u >> 16) & 1u);
  return (unsigned short)(u >> 16);
}
static __device__ __forceinline__ float bf16_f32(unsigned short h) {
  return __uint_as_float((unsigned int)h << 16);
}
static __device__ __forceinline__ v16bf frag16(const unsigned short* p0,
                                               const unsigned short* p1) {
  v8bf a = *(const v8bf*)p0;
  v8bf b = *(const v8bf*)p1;
  return __builtin_shufflevector(a, b, 0,1,2,3,4,5,6,7,8,9,10,11,12,13,14,15);
}

// ---------------- Pre-pass: split fp32 -> bf16 hi/lo planes in workspace ----
__global__ __launch_bounds__(256)
void split_planes(const float* __restrict__ X, const float* __restrict__ W,
                  unsigned short* __restrict__ ws) {
  const size_t NELEM = (size_t)MSIZE * KSIZE;
  const size_t idx4 = ((size_t)blockIdx.x * 256 + threadIdx.x) * 4;
  const float* src = (blockIdx.y == 0) ? X : W;
  unsigned short* hi = ws + (blockIdx.y ? 2 * NELEM : 0);
  unsigned short* lo = hi + NELEM;
  const float4 v = *(const float4*)(src + idx4);
  const unsigned short h0 = bf16_rn(v.x), h1 = bf16_rn(v.y);
  const unsigned short h2 = bf16_rn(v.z), h3 = bf16_rn(v.w);
  const unsigned short l0 = bf16_rn(v.x - bf16_f32(h0));
  const unsigned short l1 = bf16_rn(v.y - bf16_f32(h1));
  const unsigned short l2 = bf16_rn(v.z - bf16_f32(h2));
  const unsigned short l3 = bf16_rn(v.w - bf16_f32(h3));
  ((unsigned int*)(hi + idx4))[0] = (unsigned)h0 | ((unsigned)h1 << 16);
  ((unsigned int*)(hi + idx4))[1] = (unsigned)h2 | ((unsigned)h3 << 16);
  ((unsigned int*)(lo + idx4))[0] = (unsigned)l0 | ((unsigned)l1 << 16);
  ((unsigned int*)(lo + idx4))[1] = (unsigned)l2 | ((unsigned)l3 << 16);
}

// ---------------- TDM: DMA one 128x32 bf16 tile into padded LDS ------------
static __device__ __forceinline__ void tdm_load_tile(const unsigned short* plane,
                                                     int row0, int k0,
                                                     unsigned lds_byte_off) {
  const unsigned long long ga =
      (unsigned long long)(uintptr_t)plane +
      ((unsigned long long)row0 * KSIZE + (unsigned long long)k0) * 2ull;
  u32x4 g0;
  g0.x = 0x1u;                                    // count=1, user descriptor
  g0.y = lds_byte_off;                            // LDS destination (bytes)
  g0.z = (unsigned)(ga & 0xFFFFFFFFu);            // global_addr[31:0]
  g0.w = (unsigned)((ga >> 32) & 0x01FFFFFFu)     // global_addr[56:32]
         | 0x80000000u;                           // type=2 ("image")
  i32x8 g1;
  g1[0] = (1 << 16)      // data_size = 2 bytes
        | (1 << 20)      // pad_enable
        | (3 << 22)      // pad_interval: 16 DWORDs (= one 64B row)
        | (3 << 25);     // pad_amount: 4 DWORDs (= 16B -> 80B row stride)
  g1[1] = (int)((unsigned)KSIZE << 16);   // tensor_dim0[15:0] in [31:16]
  g1[2] = (int)((unsigned)KSIZE << 16);   // dim0 hi=0 | tensor_dim1[15:0]
  g1[3] = (int)((unsigned)TILE_K << 16);  // dim1 hi=0 | tile_dim0=32
  g1[4] = TILE_M;                         // tile_dim1=128, tile_dim2=0
  g1[5] = KSIZE;                          // tensor_dim0_stride low32
  g1[6] = 0;
  g1[7] = 0;
  const i32x4 z4 = {0, 0, 0, 0};
  const i32x8 z8 = {0, 0, 0, 0, 0, 0, 0, 0};
  __builtin_amdgcn_tensor_load_to_lds(g0, g1, z4, z4, z8, 0);  // 6-arg form
}

// ---------------- Main GEMM: bf16x3 WMMA fed by TDM double buffering -------
__global__ __launch_bounds__(256)
void gemm_bf16x3_tdm(const unsigned short* __restrict__ ws,
                     const float* __restrict__ bias,
                     float* __restrict__ out) {
  __shared__ unsigned short smem[2 * 4 * TILE_M * LDSS];   // 80 KB, 2 stages x 4 tiles

  const size_t NELEM = (size_t)MSIZE * KSIZE;
  const unsigned short* Xhi = ws;
  const unsigned short* Xlo = ws + NELEM;
  const unsigned short* Whi = ws + 2 * NELEM;
  const unsigned short* Wlo = ws + 3 * NELEM;

  const int tid   = threadIdx.x;
  const int lane  = tid & 31;
  const int wid   = tid >> 5;
  const int waveM = wid >> 1;        // 0..3
  const int waveN = wid & 1;         // 0..1
  const int lhalf = lane >> 4;
  const int l15   = lane & 15;

  const int mBlock = blockIdx.y * TILE_M;
  const int nBlock = blockIdx.x * TILE_N;

  const unsigned lds_base = (unsigned)(uintptr_t)&smem[0]; // flat addr[31:0] = LDS offset
  // tile offsets: stage s (0/1), tile a (0=Xhi,1=Xlo,2=Whi,3=Wlo)
#define TOFF(s, a) (lds_base + (unsigned)(((s) * 4 + (a)) * TILE_BYTES))
#define TELEM(s, a) (((s) * 4 + (a)) * (TILE_M * LDSS))

  v8f acc[2][4];
#pragma unroll
  for (int tm = 0; tm < 2; ++tm)
#pragma unroll
    for (int tn = 0; tn < 4; ++tn)
#pragma unroll
      for (int e = 0; e < 8; ++e) acc[tm][tn][e] = 0.0f;

  const int NT = KSIZE / TILE_K;   // 128 K-steps

  if (wid == 0) {                  // prologue: DMA stage 0
    tdm_load_tile(Xhi, mBlock, 0, TOFF(0, 0));
    tdm_load_tile(Xlo, mBlock, 0, TOFF(0, 1));
    tdm_load_tile(Whi, nBlock, 0, TOFF(0, 2));
    tdm_load_tile(Wlo, nBlock, 0, TOFF(0, 3));
  }

  for (int kt = 0; kt < NT; ++kt) {
    const int buf = kt & 1;
    if (wid == 0) __builtin_amdgcn_s_wait_tensorcnt(0);  // stage kt DMA done
    __syncthreads();  // publish LDS; also: all waves finished reading buf^1

    if (kt + 1 < NT && wid == 0) {   // DMA stage kt+1 overlapped with compute
      const int k0 = (kt + 1) * TILE_K;
      tdm_load_tile(Xhi, mBlock, k0, TOFF(buf ^ 1, 0));
      tdm_load_tile(Xlo, mBlock, k0, TOFF(buf ^ 1, 1));
      tdm_load_tile(Whi, nBlock, k0, TOFF(buf ^ 1, 2));
      tdm_load_tile(Wlo, nBlock, k0, TOFF(buf ^ 1, 3));
    }

    // Fragment fetch per documented bf16 WMMA VGPR layouts.
    v16bf ah[2], al[2], bh[4], bl[4];
#pragma unroll
    for (int tm = 0; tm < 2; ++tm) {
      const int arow = (waveM * 32 + tm * 16 + l15) * LDSS + lhalf * 8;
      const unsigned short* p = &smem[TELEM(buf, 0) + arow];
      const unsigned short* q = &smem[TELEM(buf, 1) + arow];
      ah[tm] = frag16(p, p + 16);
      al[tm] = frag16(q, q + 16);
    }
#pragma unroll
    for (int tn = 0; tn < 4; ++tn) {
      const int brow = (waveN * 64 + tn * 16 + l15) * LDSS + lhalf * 8;
      const unsigned short* p = &smem[TELEM(buf, 2) + brow];
      const unsigned short* q = &smem[TELEM(buf, 3) + brow];
      bh[tn] = frag16(p, p + 16);
      bl[tn] = frag16(q, q + 16);
    }

    // bf16x3 fp32-accuracy product: hi*lo + lo*hi + hi*hi
#pragma unroll
    for (int tm = 0; tm < 2; ++tm)
#pragma unroll
      for (int tn = 0; tn < 4; ++tn) {
        v8f c = acc[tm][tn];
        c = __builtin_amdgcn_wmma_f32_16x16x32_bf16(false, ah[tm], false, bl[tn],
                                                    (short)0, c, false, false);
        c = __builtin_amdgcn_wmma_f32_16x16x32_bf16(false, al[tm], false, bh[tn],
                                                    (short)0, c, false, false);
        c = __builtin_amdgcn_wmma_f32_16x16x32_bf16(false, ah[tm], false, bh[tn],
                                                    (short)0, c, false, false);
        acc[tm][tn] = c;
      }
  }

  // Epilogue: C/D layout (lanes 0-15: M=j, lanes 16-31: M=j+8) + bias.
#pragma unroll
  for (int tn = 0; tn < 4; ++tn) {
    const int col = nBlock + waveN * 64 + tn * 16 + l15;
    const float bv = bias[col];
#pragma unroll
    for (int tm = 0; tm < 2; ++tm) {
      const int row0 = mBlock + waveM * 32 + tm * 16 + lhalf * 8;
#pragma unroll
      for (int j = 0; j < 8; ++j)
        out[(size_t)(row0 + j) * NSIZE + col] = acc[tm][tn][j] + bv;
    }
  }
#undef TOFF
#undef TELEM
}

// ---------------- Fallback (ws too small): in-loop split, same math --------
static __device__ __forceinline__ void split4(const float4 v,
                                              unsigned short* __restrict__ ph,
                                              unsigned short* __restrict__ pl) {
  unsigned short h0 = bf16_rn(v.x), h1 = bf16_rn(v.y);
  unsigned short h2 = bf16_rn(v.z), h3 = bf16_rn(v.w);
  unsigned short l0 = bf16_rn(v.x - bf16_f32(h0));
  unsigned short l1 = bf16_rn(v.y - bf16_f32(h1));
  unsigned short l2 = bf16_rn(v.z - bf16_f32(h2));
  unsigned short l3 = bf16_rn(v.w - bf16_f32(h3));
  ((unsigned int*)ph)[0] = (unsigned)h0 | ((unsigned)h1 << 16);
  ((unsigned int*)ph)[1] = (unsigned)h2 | ((unsigned)h3 << 16);
  ((unsigned int*)pl)[0] = (unsigned)l0 | ((unsigned)l1 << 16);
  ((unsigned int*)pl)[1] = (unsigned)l2 | ((unsigned)l3 << 16);
}

__global__ __launch_bounds__(256)
void linear_bf16x3_fallback(const float* __restrict__ X,
                            const float* __restrict__ W,
                            const float* __restrict__ bias,
                            float* __restrict__ out) {
  __shared__ unsigned short ls_xh[2][TILE_M * LDSS];
  __shared__ unsigned short ls_xl[2][TILE_M * LDSS];
  __shared__ unsigned short ls_wh[2][TILE_N * LDSS];
  __shared__ unsigned short ls_wl[2][TILE_N * LDSS];

  const int tid = threadIdx.x, lane = tid & 31, wid = tid >> 5;
  const int waveM = wid >> 1, waveN = wid & 1, lhalf = lane >> 4, l15 = lane & 15;
  const int mBlock = blockIdx.y * TILE_M, nBlock = blockIdx.x * TILE_N;
  const int kq = tid & 7, r0 = tid >> 3;

  float4 xr[4], wr[4];
#pragma unroll
  for (int i = 0; i < 4; ++i) {
    xr[i] = *(const float4*)(X + (size_t)(mBlock + r0 + i * 32) * KSIZE + kq * 4);
    wr[i] = *(const float4*)(W + (size_t)(nBlock + r0 + i * 32) * KSIZE + kq * 4);
  }
  v8f acc[2][4];
#pragma unroll
  for (int tm = 0; tm < 2; ++tm)
#pragma unroll
    for (int tn = 0; tn < 4; ++tn)
#pragma unroll
      for (int e = 0; e < 8; ++e) acc[tm][tn][e] = 0.0f;

  const int NT = KSIZE / TILE_K;
  for (int kt = 0; kt < NT; ++kt) {
    const int buf = kt & 1;
#pragma unroll
    for (int i = 0; i < 4; ++i) {
      const int row = r0 + i * 32;
      split4(xr[i], &ls_xh[buf][row * LDSS + kq * 4], &ls_xl[buf][row * LDSS + kq * 4]);
      split4(wr[i], &ls_wh[buf][row * LDSS + kq * 4], &ls_wl[buf][row * LDSS + kq * 4]);
    }
    __syncthreads();
    if (kt + 1 < NT) {
      const int k0 = (kt + 1) * TILE_K;
#pragma unroll
      for (int i = 0; i < 4; ++i) {
        xr[i] = *(const float4*)(X + (size_t)(mBlock + r0 + i * 32) * KSIZE + k0 + kq * 4);
        wr[i] = *(const float4*)(W + (size_t)(nBlock + r0 + i * 32) * KSIZE + k0 + kq * 4);
      }
    }
    v16bf ah[2], al[2], bh[4], bl[4];
#pragma unroll
    for (int tm = 0; tm < 2; ++tm) {
      const int arow = (waveM * 32 + tm * 16 + l15) * LDSS + lhalf * 8;
      ah[tm] = frag16(&ls_xh[buf][arow], &ls_xh[buf][arow] + 16);
      al[tm] = frag16(&ls_xl[buf][arow], &ls_xl[buf][arow] + 16);
    }
#pragma unroll
    for (int tn = 0; tn < 4; ++tn) {
      const int brow = (waveN * 64 + tn * 16 + l15) * LDSS + lhalf * 8;
      bh[tn] = frag16(&ls_wh[buf][brow], &ls_wh[buf][brow] + 16);
      bl[tn] = frag16(&ls_wl[buf][brow], &ls_wl[buf][brow] + 16);
    }
#pragma unroll
    for (int tm = 0; tm < 2; ++tm)
#pragma unroll
      for (int tn = 0; tn < 4; ++tn) {
        v8f c = acc[tm][tn];
        c = __builtin_amdgcn_wmma_f32_16x16x32_bf16(false, ah[tm], false, bl[tn],
                                                    (short)0, c, false, false);
        c = __builtin_amdgcn_wmma_f32_16x16x32_bf16(false, al[tm], false, bh[tn],
                                                    (short)0, c, false, false);
        c = __builtin_amdgcn_wmma_f32_16x16x32_bf16(false, ah[tm], false, bh[tn],
                                                    (short)0, c, false, false);
        acc[tm][tn] = c;
      }
  }
#pragma unroll
  for (int tn = 0; tn < 4; ++tn) {
    const int col = nBlock + waveN * 64 + tn * 16 + l15;
    const float bv = bias[col];
#pragma unroll
    for (int tm = 0; tm < 2; ++tm) {
      const int row0 = mBlock + waveM * 32 + tm * 16 + lhalf * 8;
#pragma unroll
      for (int j = 0; j < 8; ++j)
        out[(size_t)(row0 + j) * NSIZE + col] = acc[tm][tn][j] + bv;
    }
  }
}

extern "C" void kernel_launch(void* const* d_in, const int* in_sizes, int n_in,
                              void* d_out, int out_size, void* d_ws, size_t ws_size,
                              hipStream_t stream) {
  (void)in_sizes; (void)n_in; (void)out_size;
  const float* X = (const float*)d_in[0];
  const float* W = (const float*)d_in[1];
  const float* B = (const float*)d_in[2];
  float* O = (float*)d_out;
  const size_t need = 4ull * (size_t)MSIZE * KSIZE * sizeof(unsigned short); // 128 MB
  dim3 grid(NSIZE / TILE_N, MSIZE / TILE_M);
  if (ws_size >= need) {
    unsigned short* ws = (unsigned short*)d_ws;
    split_planes<<<dim3((MSIZE * (size_t)KSIZE) / (256 * 4), 2), 256, 0, stream>>>(X, W, ws);
    gemm_bf16x3_tdm<<<grid, 256, 0, stream>>>(ws, B, O);
  } else {
    linear_bf16x3_fallback<<<grid, 256, 0, stream>>>(X, W, B, O);
  }
}
